// KPlexPool_49108656062698
// MI455X (gfx1250) — matrix operand order repaired and checked by reading, not compiled, not run on hardware
//
#include <hip/hip_runtime.h>

typedef __attribute__((ext_vector_type(16))) __bf16 v16bf;
typedef __attribute__((ext_vector_type(8)))  __bf16 v8bf;
typedef __attribute__((ext_vector_type(8)))  float  v8f;

#define LDS_PAD 136   // 272B row stride: 16B-aligned, conflict-free b128 LDS reads

// ---------------------------------------------------------------------------
// Utility kernels
// ---------------------------------------------------------------------------
__global__ void zero_f32(float* __restrict__ p, long n) {
    long t = (long)blockIdx.x * blockDim.x + threadIdx.x;
    if (t < n) p[t] = 0.0f;
}

__global__ void degree_count(const int* __restrict__ dst, int e, float* __restrict__ cnt) {
    int i = blockIdx.x * blockDim.x + threadIdx.x;
    if (i < e) atomicAdd(&cnt[dst[i]], 1.0f);
}

// acc[dst[e]][:] += x[src[e]][:]   (one wave per edge: 32 lanes x float4 = 512B)
__global__ void scatter_add_feat(const float* __restrict__ x, const int* __restrict__ src,
                                 const int* __restrict__ dst, int e, float* __restrict__ acc) {
    long t = (long)blockIdx.x * blockDim.x + threadIdx.x;
    long edge = t >> 5;
    int  ch   = (int)(t & 31) * 4;
    if (edge < e) {
        int s = src[edge], d = dst[edge];
        float4 v = *(const float4*)(x + (long)s * 128 + ch);
        float* o = acc + (long)d * 128 + ch;
        atomicAdd(o + 0, v.x); atomicAdd(o + 1, v.y);
        atomicAdd(o + 2, v.z); atomicAdd(o + 3, v.w);
    }
}

// acc[dst[r]][:] += x[r][:]   (identity gather; used for batch pooling)
__global__ void pool_add(const float* __restrict__ x, const int* __restrict__ dst,
                         int m, float* __restrict__ acc) {
    long t = (long)blockIdx.x * blockDim.x + threadIdx.x;
    long row = t >> 5;
    int  ch  = (int)(t & 31) * 4;
    if (row < m) {
        int d = dst[row];
        float4 v = *(const float4*)(x + row * 128 + ch);
        float* o = acc + (long)d * 128 + ch;
        atomicAdd(o + 0, v.x); atomicAdd(o + 1, v.y);
        atomicAdd(o + 2, v.z); atomicAdd(o + 3, v.w);
    }
}

// row-wise divide by max(cnt,1)  -> segment mean
__global__ void div_rows(float* __restrict__ acc, const float* __restrict__ cnt, long n) {
    long t = (long)blockIdx.x * blockDim.x + threadIdx.x;
    long row = t >> 7;
    if (row < n) {
        float c = cnt[row];
        acc[t] *= 1.0f / fmaxf(c, 1.0f);
    }
}

// ---------------------------------------------------------------------------
// Fused dual GEMM:  out = act(A1 @ W1 + A2 @ W2 + bias), all [n,128]x[128,128]
// bf16 WMMA with fp32 accumulate. Block = 256 threads = 8 waves = 128 rows.
// ---------------------------------------------------------------------------
__device__ __forceinline__ void load8_cvt(v16bf& a, int base, const float* p) {
    float4 f0 = *(const float4*)p;
    float4 f1 = *(const float4*)(p + 4);
    a[base + 0] = (__bf16)f0.x; a[base + 1] = (__bf16)f0.y;
    a[base + 2] = (__bf16)f0.z; a[base + 3] = (__bf16)f0.w;
    a[base + 4] = (__bf16)f1.x; a[base + 5] = (__bf16)f1.y;
    a[base + 6] = (__bf16)f1.z; a[base + 7] = (__bf16)f1.w;
}

__global__ __launch_bounds__(256)
void gemm_dual_wmma(const float* __restrict__ A1, const float* __restrict__ W1,
                    const float* __restrict__ A2, const float* __restrict__ W2,
                    const float* __restrict__ bias, float* __restrict__ out,
                    int n, int do_relu) {
    // Stage both weight matrices into LDS as bf16, layout [col][k] (transposed).
    __shared__ __bf16 lds[2][128][LDS_PAD];
    for (int idx = threadIdx.x; idx < 128 * 128; idx += 256) {
        int k = idx >> 7, c = idx & 127;
        lds[0][c][k] = (__bf16)W1[idx];
        lds[1][c][k] = (__bf16)W2[idx];
    }
    __syncthreads();

    const int wave = threadIdx.x >> 5;
    const int lane = threadIdx.x & 31;
    const int hs   = lane >> 4;      // K-half select
    const int lrow = lane & 15;
    const long rowBase = (long)blockIdx.x * 128 + wave * 16;
    if (rowBase >= n) return;        // n is a multiple of 16; wave-uniform branch

    v8f acc[8];
#pragma unroll
    for (int t = 0; t < 8; ++t) acc[t] = (v8f){0.f,0.f,0.f,0.f,0.f,0.f,0.f,0.f};

    const float* p1 = A1 + (rowBase + lrow) * 128;
    const float* p2 = A2 + (rowBase + lrow) * 128;

#pragma unroll
    for (int k0 = 0; k0 < 128; k0 += 32) {
        // A fragment (16x32 bf16): lanes 0-15 hold K {0..15}+k0,
        //                          lanes 16-31 hold K {16..31}+k0.
        const int kc0 = k0 + 16 * hs;
        const int kc1 = k0 + 16 * hs + 8;
        v16bf a1, a2;
        load8_cvt(a1, 0, p1 + kc0); load8_cvt(a1, 8, p1 + kc1);
        load8_cvt(a2, 0, p2 + kc0); load8_cvt(a2, 8, p2 + kc1);

        const int kB = k0 + 16 * hs;  // B fragment: lane = column, K half by lane[4]
#pragma unroll
        for (int t = 0; t < 8; ++t) {
            const __bf16* wp0 = &lds[0][t * 16 + lrow][kB];
            const __bf16* wp1 = &lds[1][t * 16 + lrow][kB];
            v8bf b0lo = *(const v8bf*)wp0;
            v8bf b0hi = *(const v8bf*)(wp0 + 8);
            v8bf b1lo = *(const v8bf*)wp1;
            v8bf b1hi = *(const v8bf*)(wp1 + 8);
            v16bf b1 = __builtin_shufflevector(b0lo, b0hi, 0,1,2,3,4,5,6,7,8,9,10,11,12,13,14,15);
            v16bf b2 = __builtin_shufflevector(b1lo, b1hi, 0,1,2,3,4,5,6,7,8,9,10,11,12,13,14,15);
            acc[t] = __builtin_amdgcn_wmma_f32_16x16x32_bf16(false, a1, false, b1,
                                                             (short)0, acc[t], false, false);
            acc[t] = __builtin_amdgcn_wmma_f32_16x16x32_bf16(false, a2, false, b2,
                                                             (short)0, acc[t], false, false);
        }
    }

    // Epilogue: C/D layout -> row = rowBase + i + 8*hs, col = t*16 + (lane&15)
#pragma unroll
    for (int t = 0; t < 8; ++t) {
        int col = t * 16 + lrow;
        float bv = bias[col];
#pragma unroll
        for (int i = 0; i < 8; ++i) {
            long row = rowBase + i + 8 * hs;
            float v = acc[t][i] + bv;
            if (do_relu) v = fmaxf(v, 0.0f);
            out[row * 128 + col] = v;
        }
    }
}

// ---------------------------------------------------------------------------
// Classifier: logits = g @ lin2_W + b, then log_softmax.  g is [64,128].
// ---------------------------------------------------------------------------
__global__ void classify_kernel(const float* __restrict__ g, const float* __restrict__ W,
                                const float* __restrict__ b, float* __restrict__ out, int B) {
    int row = threadIdx.x;
    if (row >= B) return;
    float logit[10];
#pragma unroll
    for (int c = 0; c < 10; ++c) logit[c] = b[c];
    for (int k = 0; k < 128; ++k) {
        float gv = g[row * 128 + k];
#pragma unroll
        for (int c = 0; c < 10; ++c) logit[c] += gv * W[k * 10 + c];
    }
    float m = logit[0];
#pragma unroll
    for (int c = 1; c < 10; ++c) m = fmaxf(m, logit[c]);
    float s = 0.0f;
#pragma unroll
    for (int c = 0; c < 10; ++c) s += __expf(logit[c] - m);
    float lse = __logf(s) + m;
#pragma unroll
    for (int c = 0; c < 10; ++c) out[row * 10 + c] = logit[c] - lse;
}

// ---------------------------------------------------------------------------
// Host-side orchestration
// ---------------------------------------------------------------------------
static inline int g256(long n) { return (int)((n + 255) / 256); }

extern "C" void kernel_launch(void* const* d_in, const int* in_sizes, int n_in,
                              void* d_out, int out_size, void* d_ws, size_t ws_size,
                              hipStream_t stream) {
    const float* x        = (const float*)d_in[0];
    const int*   ei       = (const int*)d_in[1];   // [2,E] src|dst
    const int*   cidx     = (const int*)d_in[3];   // [2,M] c_nodes|c_clust
    const int*   eip      = (const int*)d_in[4];   // [2,E2]
    const int*   batch_c  = (const int*)d_in[5];   // [C]
    const float* lin2_W   = (const float*)d_in[6];
    const float* lin2_b   = (const float*)d_in[7];
    const float* in_c1_Wl = (const float*)d_in[8];
    const float* in_c1_bl = (const float*)d_in[9];
    const float* in_c1_Wr = (const float*)d_in[10];
    const float* in_c2_Wl = (const float*)d_in[11];
    const float* in_c2_bl = (const float*)d_in[12];
    const float* in_c2_Wr = (const float*)d_in[13];
    const float* in_lin_W = (const float*)d_in[14];
    const float* in_lin_b = (const float*)d_in[15];
    const float* b1_c1_Wl = (const float*)d_in[16];
    const float* b1_c1_bl = (const float*)d_in[17];
    const float* b1_c1_Wr = (const float*)d_in[18];
    const float* b1_c2_Wl = (const float*)d_in[19];
    const float* b1_c2_bl = (const float*)d_in[20];
    const float* b1_c2_Wr = (const float*)d_in[21];
    const float* b1_lin_W = (const float*)d_in[22];
    const float* b1_lin_b = (const float*)d_in[23];
    float* outp = (float*)d_out;

    const int N  = in_sizes[0] / 128;
    const int E  = in_sizes[1] / 2;
    const int M  = in_sizes[3] / 2;
    const int E2 = in_sizes[4] / 2;
    const int C  = in_sizes[5];
    const int B  = out_size / 10;

    const size_t nf = (size_t)N * 128 * sizeof(float);
    const size_t cf = (size_t)C * 128 * sizeof(float);
    const size_t bf = (size_t)B * 128 * sizeof(float);
    char* ws = (char*)d_ws;
    float* bufA  = (float*)(ws);                 // agg scratch, then h
    float* bufX1 = (float*)(ws + nf);
    float* bufX2 = (float*)(ws + 2 * nf);
    float* hcA   = (float*)(ws + 3 * nf);        // cluster agg scratch, then hc_out
    float* hcX   = (float*)(ws + 3 * nf + cf);   // hc (cover pooled)
    float* hcY1  = (float*)(ws + 3 * nf + 2 * cf);
    float* hcY2  = (float*)(ws + 3 * nf + 3 * cf);
    float* gbuf  = (float*)(ws + 3 * nf + 4 * cf);
    float* cntN  = (float*)(ws + 3 * nf + 4 * cf + bf);
    float* cntC  = (float*)((char*)cntN + (size_t)N * sizeof(float));
    float* cntB  = (float*)((char*)cntC + (size_t)C * sizeof(float));

    const int*  src = ei,     *dst = ei + E;
    const int* srcp = eip,   *dstp = eip + E2;
    const int* cnod = cidx,  *ccls = cidx + M;
    const long NF = (long)N * 128, CF = (long)C * 128, BF = (long)B * 128;

    // ---- node-level degrees
    zero_f32<<<g256(N), 256, 0, stream>>>(cntN, N);
    degree_count<<<g256(E), 256, 0, stream>>>(dst, E, cntN);

    // ---- conv1: x1 = relu(mean(x[src]->dst)@Wl + x@Wr + bl)
    zero_f32<<<g256(NF), 256, 0, stream>>>(bufA, NF);
    scatter_add_feat<<<g256((long)E * 32), 256, 0, stream>>>(x, src, dst, E, bufA);
    div_rows<<<g256(NF), 256, 0, stream>>>(bufA, cntN, N);
    gemm_dual_wmma<<<(N + 127) / 128, 256, 0, stream>>>(bufA, in_c1_Wl, x, in_c1_Wr,
                                                        in_c1_bl, bufX1, N, 1);
    // ---- conv2
    zero_f32<<<g256(NF), 256, 0, stream>>>(bufA, NF);
    scatter_add_feat<<<g256((long)E * 32), 256, 0, stream>>>(bufX1, src, dst, E, bufA);
    div_rows<<<g256(NF), 256, 0, stream>>>(bufA, cntN, N);
    gemm_dual_wmma<<<(N + 127) / 128, 256, 0, stream>>>(bufA, in_c2_Wl, bufX1, in_c2_Wr,
                                                        in_c2_bl, bufX2, N, 1);
    // ---- concat-linear: h = relu(x1@Wtop + x2@Wbot + b)  (into bufA)
    gemm_dual_wmma<<<(N + 127) / 128, 256, 0, stream>>>(bufX1, in_lin_W, bufX2,
                                                        in_lin_W + 128 * 128,
                                                        in_lin_b, bufA, N, 1);
    // ---- cover pool: hc = seg_mean(h[c_nodes], c_clust, C)
    zero_f32<<<g256(C), 256, 0, stream>>>(cntC, C);
    degree_count<<<g256(M), 256, 0, stream>>>(ccls, M, cntC);
    zero_f32<<<g256(CF), 256, 0, stream>>>(hcX, CF);
    scatter_add_feat<<<g256((long)M * 32), 256, 0, stream>>>(bufA, cnod, ccls, M, hcX);
    div_rows<<<g256(CF), 256, 0, stream>>>(hcX, cntC, C);

    // ---- cluster-level degrees
    zero_f32<<<g256(C), 256, 0, stream>>>(cntC, C);
    degree_count<<<g256(E2), 256, 0, stream>>>(dstp, E2, cntC);

    // ---- b1 conv1
    zero_f32<<<g256(CF), 256, 0, stream>>>(hcA, CF);
    scatter_add_feat<<<g256((long)E2 * 32), 256, 0, stream>>>(hcX, srcp, dstp, E2, hcA);
    div_rows<<<g256(CF), 256, 0, stream>>>(hcA, cntC, C);
    gemm_dual_wmma<<<(C + 127) / 128, 256, 0, stream>>>(hcA, b1_c1_Wl, hcX, b1_c1_Wr,
                                                        b1_c1_bl, hcY1, C, 1);
    // ---- b1 conv2
    zero_f32<<<g256(CF), 256, 0, stream>>>(hcA, CF);
    scatter_add_feat<<<g256((long)E2 * 32), 256, 0, stream>>>(hcY1, srcp, dstp, E2, hcA);
    div_rows<<<g256(CF), 256, 0, stream>>>(hcA, cntC, C);
    gemm_dual_wmma<<<(C + 127) / 128, 256, 0, stream>>>(hcA, b1_c2_Wl, hcY1, b1_c2_Wr,
                                                        b1_c2_bl, hcY2, C, 1);
    // ---- b1 concat-linear (into hcA)
    gemm_dual_wmma<<<(C + 127) / 128, 256, 0, stream>>>(hcY1, b1_lin_W, hcY2,
                                                        b1_lin_W + 128 * 128,
                                                        b1_lin_b, hcA, C, 1);
    // ---- global mean pool over graphs
    zero_f32<<<g256(B), 256, 0, stream>>>(cntB, B);
    degree_count<<<g256(C), 256, 0, stream>>>(batch_c, C, cntB);
    zero_f32<<<g256(BF), 256, 0, stream>>>(gbuf, BF);
    pool_add<<<g256((long)C * 32), 256, 0, stream>>>(hcA, batch_c, C, gbuf);
    div_rows<<<g256(BF), 256, 0, stream>>>(gbuf, cntB, B);

    // ---- classifier + log_softmax
    classify_kernel<<<1, 64, 0, stream>>>(gbuf, lin2_W, lin2_b, outp, B);
}
